// SocialModel_59399397703790
// MI455X (gfx1250) — matrix operand order repaired and checked by reading, not compiled, not run on hardware
//
#include <hip/hip_runtime.h>

// ---------------- types ----------------
typedef __bf16 bf16_t;
typedef __attribute__((ext_vector_type(16))) __bf16   v16bf;
typedef __attribute__((ext_vector_type(8)))  float    v8f;
typedef __attribute__((ext_vector_type(4)))  float    f32x4;
typedef __attribute__((ext_vector_type(4)))  unsigned u32x4;

union ABu { v16bf v; u32x4 q[2]; };

__device__ __forceinline__ v8f wmma_bf16(v16bf a, v16bf b, v8f c) {
    // emits v_wmma_f32_16x16x32_bf16
    return __builtin_amdgcn_wmma_f32_16x16x32_bf16(false, a, false, b, (short)0, c, false, false);
}

// Problem constants
#define TT   20
#define NN   512
#define G2   16
#define RR   128
#define EE   64
#define OUTW 5

// Workspace byte offsets (d_ws)
#define OFF_HT   0u         // hT  bf16 [128][512]          131072 B
#define OFF_HB   131072u    // h   bf16 [512][128]          131072 B
#define OFF_HF   262144u    // h   f32  [512][128]          262144 B
#define OFF_C    524288u    // c   f32  [512][128]          262144 B
#define OFF_SOC  786432u    // social bf16 [512][2048]      2097152 B
#define OFF_X    2883584u   // x   bf16 [512][128]          131072 B
#define OFF_Z    3014656u   // z   f32  [512][512]          1048576 B
#define OFF_WT   4063232u   // W_tensorT bf16 [64][2048]    262144 B
#define OFF_WB   4325376u   // [K;U]T    bf16 [512][256]    262144 B

// ---------------- prep kernels (once per launch) ----------------
__global__ void k_zero(unsigned* __restrict__ p) {
    int id = blockIdx.x * 256 + threadIdx.x;     // 196608 words = hT,hB,hF,c
    if (id < 196608) p[id] = 0u;
}

__global__ void k_prep_wt(const float* __restrict__ Wt, bf16_t* __restrict__ WtT) {
    int id = blockIdx.x * 256 + threadIdx.x;     // 2048*64
    if (id >= 2048 * 64) return;
    int k = id >> 6, e = id & 63;                // Wt is [2048][64] row-major
    WtT[(size_t)e * 2048 + k] = (bf16_t)Wt[id];
}

__global__ void k_prep_wb(const float* __restrict__ Kl, const float* __restrict__ Ul,
                          bf16_t* __restrict__ WbT) {
    int id = blockIdx.x * 256 + threadIdx.x;     // 512*256, id = col*256 + k
    if (id >= 512 * 256) return;
    int col = id >> 8, k = id & 255;
    float v = (k < 128) ? Kl[(size_t)k * 512 + col] : Ul[(size_t)(k - 128) * 512 + col];
    WbT[id] = (bf16_t)v;
}

// ---------------- step kernel 1: social pooling GEMM ----------------
// C[(n*16+g)][r] = sum_m grid[n][m][g] * h[m][r]   -> social bf16 [512][2048]
// Block: 256 thr = 8 waves; wave w: n = blk*4 + (w>>1), cols = (w&1)*64 .. +64
// 64-m LDS rounds; grid staging double-buffered in regs; all 8 B tiles of a
// round issued in one clause before the A builds so LDS/cvt work hides them.
__global__ __launch_bounds__(256) void k_social(const float* __restrict__ grids_t,  // [512][512][16]
                                                const bf16_t* __restrict__ hT,      // [128][512]
                                                bf16_t* __restrict__ social) {
    __shared__ float lds[4 * 1088];              // 4 agents x 64 m x (16 g, pad 17)
    const int tid  = threadIdx.x;
    const int wave = tid >> 5, lane = tid & 31;
    const int l16  = lane & 15, hl = lane >> 4;
    const int n_i  = wave >> 1;
    const int colbase = (wave & 1) * 64;
    const int n = blockIdx.x * 4 + n_i;

    v8f acc[4] = {{}, {}, {}, {}};

    const int ni  = tid >> 6;                    // staging agent index
    const int idx = (tid & 63) * 16;             // staging element base (of 1024)
    const float* sbase = grids_t + (size_t)(blockIdx.x * 4 + ni) * 8192 + idx;

    // prologue: stage round 0 loads into registers
    f32x4 s0 = ((const f32x4*)sbase)[0];
    f32x4 s1 = ((const f32x4*)sbase)[1];
    f32x4 s2 = ((const f32x4*)sbase)[2];
    f32x4 s3 = ((const f32x4*)sbase)[3];

    for (int kb = 0; kb < 512; kb += 64) {
        __syncthreads();
        {   // write staged registers to LDS (transposed, padded stride 17)
            float tmp[16] = {s0.x, s0.y, s0.z, s0.w, s1.x, s1.y, s1.z, s1.w,
                             s2.x, s2.y, s2.z, s2.w, s3.x, s3.y, s3.z, s3.w};
#pragma unroll
            for (int j = 0; j < 16; ++j) {
                int m = (idx + j) >> 4, g = (idx + j) & 15;
                lds[ni * 1088 + m * 17 + g] = tmp[j];
            }
        }
        __syncthreads();
        // issue next round's HBM staging loads now; consumed after this round's WMMAs
        if (kb + 64 < 512) {
            const float* src = sbase + (size_t)(kb + 64) * 16;
            s0 = ((const f32x4*)src)[0];
            s1 = ((const f32x4*)src)[1];
            s2 = ((const f32x4*)src)[2];
            s3 = ((const f32x4*)src)[3];
            __builtin_prefetch(src + 1024, 0, 1);     // global_prefetch_b8 (round +2)
        }
        // issue ALL 8 B tiles for this round in one clause (distinct live ranges)
        const bf16_t* bb = hT + (size_t)(colbase + l16) * 512 + kb + 16 * hl;
        v16bf b0 = *(const v16bf*)(bb);
        v16bf b1 = *(const v16bf*)(bb + 16 * 512);
        v16bf b2 = *(const v16bf*)(bb + 32 * 512);
        v16bf b3 = *(const v16bf*)(bb + 48 * 512);
        v16bf b4 = *(const v16bf*)(bb + 32);
        v16bf b5 = *(const v16bf*)(bb + 32 + 16 * 512);
        v16bf b6 = *(const v16bf*)(bb + 32 + 32 * 512);
        v16bf b7 = *(const v16bf*)(bb + 32 + 48 * 512);
        // A tile kh=0 from LDS (overlaps the global loads above)
        ABu a0;
#pragma unroll
        for (int i = 0; i < 8; ++i) {
            int kA = 16 * hl + i;
            int kB = 8 + 16 * hl + i;
            a0.v[i]     = (bf16_t)lds[n_i * 1088 + kA * 17 + l16];
            a0.v[8 + i] = (bf16_t)lds[n_i * 1088 + kB * 17 + l16];
        }
        acc[0] = wmma_bf16(a0.v, b0, acc[0]);
        acc[1] = wmma_bf16(a0.v, b1, acc[1]);
        acc[2] = wmma_bf16(a0.v, b2, acc[2]);
        acc[3] = wmma_bf16(a0.v, b3, acc[3]);
        // A tile kh=1
        ABu a1;
#pragma unroll
        for (int i = 0; i < 8; ++i) {
            int kA = 32 + 16 * hl + i;
            int kB = 40 + 16 * hl + i;
            a1.v[i]     = (bf16_t)lds[n_i * 1088 + kA * 17 + l16];
            a1.v[8 + i] = (bf16_t)lds[n_i * 1088 + kB * 17 + l16];
        }
        acc[0] = wmma_bf16(a1.v, b4, acc[0]);
        acc[1] = wmma_bf16(a1.v, b5, acc[1]);
        acc[2] = wmma_bf16(a1.v, b6, acc[2]);
        acc[3] = wmma_bf16(a1.v, b7, acc[3]);
    }
    bf16_t* outn = social + (size_t)n * 2048;    // [g*128 + r]
#pragma unroll
    for (int ct = 0; ct < 4; ++ct) {
        int col = colbase + ct * 16 + l16;
#pragma unroll
        for (int j = 0; j < 8; ++j) {
            int g = j + 8 * hl;
            outn[(size_t)g * 128 + col] = (bf16_t)acc[ct][j];
        }
    }
}

// ---------------- step kernel 2: tensor embedding GEMM + ReLU ----------------
// ten_emb = relu(social[512,2048] @ W_tensor[2048,64] + b) -> x[:,64:128)
// 1-deep software pipeline: step k+64 loads issued before step k WMMAs.
__global__ __launch_bounds__(128) void k_tensor_emb(const bf16_t* __restrict__ social,
                                                    const bf16_t* __restrict__ WtT,
                                                    const float* __restrict__ bt,
                                                    bf16_t* __restrict__ x) {
    const int wave = threadIdx.x >> 5, lane = threadIdx.x & 31;
    const int l16 = lane & 15, hl = lane >> 4;
    const int gw = blockIdx.x * 4 + wave;        // 128 waves: 32 m-tiles x 4 e-tiles
    const int mt = gw >> 2, et = gw & 3;
    const int row = mt * 16 + l16;
    const int col = et * 16 + l16;
    const bf16_t* arow = social + (size_t)row * 2048;
    const bf16_t* brow = WtT + (size_t)col * 2048;

    auto ld = [&](int k0, ABu& a0, ABu& a1, v16bf& b0, v16bf& b1) {
        a0.q[0] = *(const u32x4*)(arow + k0 + 16 * hl);
        a0.q[1] = *(const u32x4*)(arow + k0 + 8 + 16 * hl);
        a1.q[0] = *(const u32x4*)(arow + k0 + 32 + 16 * hl);
        a1.q[1] = *(const u32x4*)(arow + k0 + 40 + 16 * hl);
        b0 = *(const v16bf*)(brow + k0 + 16 * hl);
        b1 = *(const v16bf*)(brow + k0 + 32 + 16 * hl);
    };

    v8f acc = {};
    ABu a0, a1;
    v16bf b0, b1;
    ld(0, a0, a1, b0, b1);
#pragma unroll 2
    for (int k0 = 0; k0 < 2048; k0 += 64) {
        ABu na0, na1;
        v16bf nb0, nb1;
        if (k0 + 64 < 2048) ld(k0 + 64, na0, na1, nb0, nb1);
        acc = wmma_bf16(a0.v, b0, acc);
        acc = wmma_bf16(a1.v, b1, acc);
        if (k0 + 64 < 2048) { a0 = na0; a1 = na1; b0 = nb0; b1 = nb1; }
    }
    float bias = bt[col];
#pragma unroll
    for (int j = 0; j < 8; ++j) {
        int nn = mt * 16 + j + 8 * hl;
        float v = acc[j] + bias;
        x[(size_t)nn * 128 + 64 + col] = (bf16_t)fmaxf(v, 0.f);
    }
}

// ---------------- step kernel 3: input embedding (tiny) ----------------
__global__ void k_input_emb(const float* __restrict__ pos, const float* __restrict__ Ws,
                            const float* __restrict__ bs, bf16_t* __restrict__ x) {
    int id = blockIdx.x * 256 + threadIdx.x;     // 512*64
    if (id >= 512 * 64) return;
    int n = id >> 6, e = id & 63;
    float v = pos[n * 2] * Ws[e] + pos[n * 2 + 1] * Ws[64 + e] + bs[e];
    x[(size_t)n * 128 + e] = (bf16_t)fmaxf(v, 0.f);
}

// ---------------- step kernel 4: z = [x|h] @ [K;U] + b ----------------
// Fully unrolled K=256 with 1-deep pipeline.
__global__ __launch_bounds__(256) void k_zmat(const bf16_t* __restrict__ x,
                                              const bf16_t* __restrict__ h,
                                              const bf16_t* __restrict__ WbT,
                                              const float* __restrict__ bl,
                                              float* __restrict__ z) {
    const int wave = threadIdx.x >> 5, lane = threadIdx.x & 31;
    const int l16 = lane & 15, hl = lane >> 4;
    const int gw = blockIdx.x * 8 + wave;        // 256 waves: 32 m-tiles x 8 col64-tiles
    const int mt = gw >> 3, nt = gw & 7;
    const int row = mt * 16 + l16;
    const int colbase = nt * 64;

    const bf16_t* xr = x + (size_t)row * 128;
    const bf16_t* hr = h + (size_t)row * 128;

    auto ldA = [&](int k0, ABu& a) {
        const bf16_t* p = (k0 < 128) ? (xr + k0) : (hr + (k0 - 128));
        a.q[0] = *(const u32x4*)(p + 16 * hl);
        a.q[1] = *(const u32x4*)(p + 8 + 16 * hl);
    };
    auto ldB = [&](int k0, v16bf& b0, v16bf& b1, v16bf& b2, v16bf& b3) {
        const bf16_t* bb = WbT + (size_t)(colbase + l16) * 256 + k0 + 16 * hl;
        b0 = *(const v16bf*)(bb);
        b1 = *(const v16bf*)(bb + 16 * 256);
        b2 = *(const v16bf*)(bb + 32 * 256);
        b3 = *(const v16bf*)(bb + 48 * 256);
    };

    v8f acc[4] = {{}, {}, {}, {}};
    ABu a;
    v16bf b0, b1, b2, b3;
    ldA(0, a);
    ldB(0, b0, b1, b2, b3);
#pragma unroll
    for (int k0 = 0; k0 < 256; k0 += 32) {
        ABu an;
        v16bf n0, n1, n2, n3;
        if (k0 + 32 < 256) { ldA(k0 + 32, an); ldB(k0 + 32, n0, n1, n2, n3); }
        acc[0] = wmma_bf16(a.v, b0, acc[0]);
        acc[1] = wmma_bf16(a.v, b1, acc[1]);
        acc[2] = wmma_bf16(a.v, b2, acc[2]);
        acc[3] = wmma_bf16(a.v, b3, acc[3]);
        if (k0 + 32 < 256) { a = an; b0 = n0; b1 = n1; b2 = n2; b3 = n3; }
    }
#pragma unroll
    for (int ct = 0; ct < 4; ++ct) {
        int col = colbase + ct * 16 + l16;
        float bias = bl[col];
#pragma unroll
        for (int j = 0; j < 8; ++j) {
            int nn = mt * 16 + j + 8 * hl;
            z[(size_t)nn * 512 + col] = acc[ct][j] + bias;
        }
    }
}

// ---------------- step kernel 5: LSTM gates / state update ----------------
__global__ void k_gates(const float* __restrict__ z, float* __restrict__ c,
                        float* __restrict__ hf, bf16_t* __restrict__ hb,
                        bf16_t* __restrict__ hT) {
    int id = blockIdx.x * 256 + threadIdx.x;     // 512*128
    if (id >= 512 * 128) return;
    int n = id >> 7, r = id & 127;
    const float* zr = z + (size_t)n * 512;
    float i_ = 1.f / (1.f + __expf(-zr[r]));
    float f_ = 1.f / (1.f + __expf(-zr[128 + r]));
    float g_ = tanhf(zr[256 + r]);
    float o_ = 1.f / (1.f + __expf(-zr[384 + r]));
    float cn = f_ * c[id] + i_ * g_;
    c[id] = cn;
    float hn = o_ * tanhf(cn);
    hf[id] = hn;
    hb[id] = (bf16_t)hn;
    hT[(size_t)r * 512 + n] = (bf16_t)hn;
}

// ---------------- step kernel 6: output projection (5 wide) ----------------
__global__ void k_out(const float* __restrict__ hf, const float* __restrict__ Wo,
                      const float* __restrict__ bo, float* __restrict__ out) {
    int id = blockIdx.x * 256 + threadIdx.x;     // 512*5
    if (id >= 512 * OUTW) return;
    int n = id / OUTW, j = id % OUTW;
    const float* hr = hf + (size_t)n * RR;
    float s = bo[j];
#pragma unroll 4
    for (int r = 0; r < RR; ++r) s += hr[r] * Wo[r * OUTW + j];
    out[(size_t)n * OUTW + j] = s;
}

// ---------------- driver ----------------
extern "C" void kernel_launch(void* const* d_in, const int* in_sizes, int n_in,
                              void* d_out, int out_size, void* d_ws, size_t ws_size,
                              hipStream_t stream) {
    const float* frame = (const float*)d_in[0];   // [20,512,2]
    const float* grids = (const float*)d_in[1];   // [20,512,512,16]
    const float* Ws    = (const float*)d_in[2];   // [2,64]
    const float* bs    = (const float*)d_in[3];   // [64]
    const float* Wt    = (const float*)d_in[4];   // [2048,64]
    const float* bt    = (const float*)d_in[5];   // [64]
    const float* Kl    = (const float*)d_in[6];   // [128,512]
    const float* Ul    = (const float*)d_in[7];   // [128,512]
    const float* bl    = (const float*)d_in[8];   // [512]
    const float* Wo    = (const float*)d_in[9];   // [128,5]
    const float* bo    = (const float*)d_in[10];  // [5]
    float* out = (float*)d_out;                   // [20,512,5]

    char* ws = (char*)d_ws;
    bf16_t* hT  = (bf16_t*)(ws + OFF_HT);
    bf16_t* hb  = (bf16_t*)(ws + OFF_HB);
    float*  hf  = (float*) (ws + OFF_HF);
    float*  cst = (float*) (ws + OFF_C);
    bf16_t* soc = (bf16_t*)(ws + OFF_SOC);
    bf16_t* xb  = (bf16_t*)(ws + OFF_X);
    float*  zb  = (float*) (ws + OFF_Z);
    bf16_t* WtT = (bf16_t*)(ws + OFF_WT);
    bf16_t* WbT = (bf16_t*)(ws + OFF_WB);

    // one-time (per launch) state zero + weight transposes to bf16
    k_zero   <<<768, 256, 0, stream>>>((unsigned*)ws);
    k_prep_wt<<<512, 256, 0, stream>>>(Wt, WtT);
    k_prep_wb<<<512, 256, 0, stream>>>(Kl, Ul, WbT);

    for (int t = 0; t < TT; ++t) {
        k_social    <<<128, 256, 0, stream>>>(grids + (size_t)t * NN * NN * G2, hT, soc);
        k_tensor_emb<<<32, 128, 0, stream>>>(soc, WtT, bt, xb);
        k_input_emb <<<128, 256, 0, stream>>>(frame + (size_t)t * NN * 2, Ws, bs, xb);
        k_zmat      <<<32, 256, 0, stream>>>(xb, hb, WbT, bl, zb);
        k_gates     <<<256, 256, 0, stream>>>(zb, cst, hf, hb, hT);
        k_out       <<<10, 256, 0, stream>>>(hf, Wo, bo, out + (size_t)t * NN * OUTW);
    }
}
